// FullLSTM_17540646437774
// MI455X (gfx1250) — matrix operand order repaired and checked by reading, not compiled, not run on hardware
//
#include <hip/hip_runtime.h>

// ---------------------------------------------------------------------------
// LSTM: B=32, T=1024, I=H=1024.
//   Pre-split: x, w_ih, w_hh -> hi/lo bf16 planes (one-time, memory-bound).
//   Phase 1:   xp = x @ w_ih^T + (b_ih+b_hh)        (split-bf16 WMMA GEMM)
//   Phase 2:   persistent kernel, gate-grouped tiles: each block owns a
//              16-wide j range and all 8 tiles (2 batch halves x 4 gate
//              groups) feeding it; reduce + cell update are block-local in
//              LDS; ONE grid barrier per step (h visibility only).
//              w_hh-hi fragments are register-cached across all 1024 steps.
// All matmuls: v_wmma_f32_16x16x32_bf16, f32 accumulation; split-bf16
// (hi+lo) operands give ~fp32-grade products at 3 WMMAs per tile.
// ---------------------------------------------------------------------------

#define BB   32
#define TT   1024
#define II   1024
#define HH   1024
#define G4   4096          // 4*H

#define REC_BLOCKS  64     // persistent-kernel grid (co-resident)
#define REC_THREADS 1024   // 32 waves (wave32) = one full WGP workgroup
#define THREADS     256    // helper kernels

typedef __attribute__((ext_vector_type(8)))  float          v8f;
typedef __attribute__((ext_vector_type(16))) __bf16         v16bf;
typedef __attribute__((ext_vector_type(8)))  __bf16         v8bf;
typedef __attribute__((ext_vector_type(8)))  unsigned short v8us;

__device__ __forceinline__ v8f wmma_bf16(v16bf a, v16bf b, v8f c) {
    // (neg_a, A, neg_b, B, c_mod, C, reuse_a, reuse_b)
    return __builtin_amdgcn_wmma_f32_16x16x32_bf16(false, a, false, b,
                                                   (short)0, c, false, false);
}

// A fragment (16x32 bf16): lane half selects K 0..15 vs 16..31 (contiguous).
__device__ __forceinline__ v16bf load_frag_A(const __bf16* row, int kc, int sA) {
    v8bf p0 = *(const v8bf*)(row + kc + sA);
    v8bf p1 = *(const v8bf*)(row + kc + sA + 8);
    return __builtin_shufflevector(p0, p1, 0,1,2,3,4,5,6,7,8,9,10,11,12,13,14,15);
}

// B fragment (32x16 bf16): lane half selects K 0..15 vs 16..31 (contiguous).
__device__ __forceinline__ v16bf load_frag_B(const __bf16* row, int kc, int sB) {
    v8bf p0 = *(const v8bf*)(row + kc + sB);
    v8bf p1 = *(const v8bf*)(row + kc + sB + 8);
    return __builtin_shufflevector(p0, p1, 0,1,2,3,4,5,6,7,8,9,10,11,12,13,14,15);
}

__device__ __forceinline__ unsigned short bf16_rne_bits(float f) {
    unsigned u = __builtin_bit_cast(unsigned, f);
    return (unsigned short)((u + 0x7FFFu + ((u >> 16) & 1u)) >> 16);
}

__device__ __forceinline__ float sigmoidf_dev(float x) {
    return 1.0f / (1.0f + expf(-x));
}

// Monotonic-counter grid barrier (agent scope); counter zeroed each call.
__device__ __forceinline__ void grid_barrier(unsigned* counter, unsigned epoch,
                                             unsigned nblocks) {
    __syncthreads();
    if (threadIdx.x == 0) {
        __threadfence();
        __hip_atomic_fetch_add(counter, 1u, __ATOMIC_RELEASE, __HIP_MEMORY_SCOPE_AGENT);
        const unsigned target = epoch * nblocks;
        while (__hip_atomic_load(counter, __ATOMIC_ACQUIRE, __HIP_MEMORY_SCOPE_AGENT) < target) {
            __builtin_amdgcn_s_sleep(1);
        }
        __threadfence();
    }
    __syncthreads();
}

// ---------------------------------------------------------------------------
// One-time split: f32 array -> hi/lo bf16 planes (RNE hi, RNE residual lo).
// ---------------------------------------------------------------------------
__global__ __launch_bounds__(THREADS) void split_kernel(
    const float* __restrict__ src, __bf16* __restrict__ hi,
    __bf16* __restrict__ lo, int n8)
{
    int i = blockIdx.x * blockDim.x + threadIdx.x;
    if (i >= n8) return;
    const float4* s = (const float4*)src + (size_t)i * 2;
    float4 f0 = s[0], f1 = s[1];
    float f[8] = {f0.x, f0.y, f0.z, f0.w, f1.x, f1.y, f1.z, f1.w};
    v8us hb, lb;
#pragma unroll
    for (int j = 0; j < 8; ++j) {
        unsigned short hbits = bf16_rne_bits(f[j]);
        float hf = __builtin_bit_cast(float, (unsigned)hbits << 16);
        hb[j] = hbits;
        lb[j] = bf16_rne_bits(f[j] - hf);
    }
    *(v8bf*)(hi + (size_t)i * 8) = __builtin_bit_cast(v8bf, hb);
    *(v8bf*)(lo + (size_t)i * 8) = __builtin_bit_cast(v8bf, lb);
}

// ---------------------------------------------------------------------------
// Init: zero c (f32), h hi/lo (bf16), barrier counter.
// ---------------------------------------------------------------------------
__global__ void lstm_init(float* __restrict__ c, __bf16* __restrict__ h_hi,
                          __bf16* __restrict__ h_lo, unsigned* __restrict__ counter) {
    int i = blockIdx.x * blockDim.x + threadIdx.x;
    if (i < BB * HH) {
        c[i] = 0.0f;
        h_hi[i] = __builtin_bit_cast(__bf16, (unsigned short)0);
        h_lo[i] = __builtin_bit_cast(__bf16, (unsigned short)0);
    }
    if (i == 0) *counter = 0u;
}

// ---------------------------------------------------------------------------
// Phase 1: xp = x @ w_ih^T + (b_ih + b_hh), from pre-split bf16 planes.
// Block tile 128x64; 8 waves; wave = 16 rows x 64 cols (4 accumulators).
// ---------------------------------------------------------------------------
__global__ __launch_bounds__(THREADS) void xproj_kernel(
    const __bf16* __restrict__ x_hi, const __bf16* __restrict__ x_lo,
    const __bf16* __restrict__ w_hi, const __bf16* __restrict__ w_lo,
    const float* __restrict__ b_ih, const float* __restrict__ b_hh,
    float* __restrict__ xp)
{
    const int lane  = threadIdx.x & 31;
    const int wave  = threadIdx.x >> 5;
    const int row0  = blockIdx.y * 128 + wave * 16;
    const int col0  = blockIdx.x * 64;
    const int laneM = lane & 15;
    const int half  = lane >> 4;
    const int sA    = half * 16;
    const int sB    = half * 16;

    v8f acc[4] = {};
    const __bf16* xh = x_hi + (size_t)(row0 + laneM) * II;
    const __bf16* xl = x_lo + (size_t)(row0 + laneM) * II;

    for (int kc = 0; kc < II; kc += 32) {
        const v16bf ah = load_frag_A(xh, kc, sA);
        const v16bf al = load_frag_A(xl, kc, sA);
#pragma unroll
        for (int ns = 0; ns < 4; ++ns) {
            const size_t wroff = (size_t)(col0 + ns * 16 + laneM) * II;
            const v16bf bh = load_frag_B(w_hi + wroff, kc, sB);
            const v16bf bl = load_frag_B(w_lo + wroff, kc, sB);
            acc[ns] = wmma_bf16(al, bh, acc[ns]);
            acc[ns] = wmma_bf16(ah, bl, acc[ns]);
            acc[ns] = wmma_bf16(ah, bh, acc[ns]);
        }
    }

#pragma unroll
    for (int ns = 0; ns < 4; ++ns) {
        const int col = col0 + ns * 16 + laneM;
        const float bias = b_ih[col] + b_hh[col];
#pragma unroll
        for (int r = 0; r < 8; ++r) {
            const int row = row0 + half * 8 + r;   // C layout: M = r + 8*half
            xp[(size_t)row * G4 + col] = acc[ns][r] + bias;
        }
    }
}

// ---------------------------------------------------------------------------
// Phase 2: persistent recurrent kernel, gate-grouped mapping.
// 64 blocks x 32 waves. Block bid owns j in [bid*16, bid*16+16) and its
// 8 tiles: tl = mhalf*4 + ggrp (batch half x gate group {i,f,g,o}).
// Wave w: tile tl = w>>2, K quarter kPart = w&3 (K=1024 split 4 ways).
// B-hi fragments (w_hh) register-cached across all timesteps.
// Reduce + cell update block-local via LDS; one grid barrier per step.
// ---------------------------------------------------------------------------
__global__ __launch_bounds__(REC_THREADS) void lstm_recurrent(
    const float* __restrict__ xp,
    const __bf16* __restrict__ whh_hi, const __bf16* __restrict__ whh_lo,
    float* __restrict__ out,
    __bf16* __restrict__ h_hi, __bf16* __restrict__ h_lo,
    float* __restrict__ c, unsigned* __restrict__ counter)
{
    __shared__ float red[32 * 256];   // 32 waves x 16x16 partials = 32 KB

    const int lane  = threadIdx.x & 31;
    const int wave  = threadIdx.x >> 5;
    const int tl    = wave >> 2;            // 0..7 = mhalf*4 + ggrp
    const int kPart = wave & 3;             // K quarter
    const int mhalf = tl >> 2;              // batch half
    const int ggrp  = tl & 3;               // gate group (i,f,g,o)
    const int j0    = blockIdx.x * 16;      // owned j range
    const int m0    = mhalf * 16;
    const int n0    = ggrp * HH + j0;       // gate-column base
    const int laneM = lane & 15;
    const int half  = lane >> 4;
    const int sA    = half * 16;
    const int sB    = half * 16;
    const int kBase = kPart * 256;

    const __bf16* hh = h_hi + (size_t)(m0 + laneM) * HH;
    const __bf16* hl = h_lo + (size_t)(m0 + laneM) * HH;
    const __bf16* wh = whh_hi + (size_t)(n0 + laneM) * HH;
    const __bf16* wl = whh_lo + (size_t)(n0 + laneM) * HH;

    // Register-cache timestep-invariant B-hi fragments (8 x 8 = 64 VGPRs).
    v16bf bhf[8];
#pragma unroll
    for (int kk = 0; kk < 8; ++kk)
        bhf[kk] = load_frag_B(wh, kBase + kk * 32, sB);

    unsigned epoch = 0;
    for (int t = 0; t < TT; ++t) {
        // ---- GEMM: A (h) streamed from L2-hot 128KB, B-hi from registers ----
        v8f acc = {};
#pragma unroll
        for (int kk = 0; kk < 8; ++kk) {
            const int kc = kBase + kk * 32;
            const v16bf ah = load_frag_A(hh, kc, sA);
            const v16bf al = load_frag_A(hl, kc, sA);
            const v16bf bl = load_frag_B(wl, kc, sB);
            acc = wmma_bf16(al, bhf[kk], acc);
            acc = wmma_bf16(ah, bl, acc);
            acc = wmma_bf16(ah, bhf[kk], acc);
        }
#pragma unroll
        for (int r = 0; r < 8; ++r)
            red[wave * 256 + (half * 8 + r) * 16 + laneM] = acc[r];
        __syncthreads();

        // ---- block-local reduce + LSTM cell update: 512 elements (b, jj) ----
        if (threadIdx.x < BB * 16) {
            const int b  = threadIdx.x >> 4;        // batch 0..31
            const int jj = threadIdx.x & 15;        // col within j range
            const int e  = (b & 15) * 16 + jj;      // element in 16x16 tile
            const int mh = b >> 4;
            float gv[4];
#pragma unroll
            for (int g = 0; g < 4; ++g) {
                const int wbase = (mh * 4 + g) * 4; // first wave of that tile
                float s = red[(wbase + 0) * 256 + e] + red[(wbase + 1) * 256 + e]
                        + red[(wbase + 2) * 256 + e] + red[(wbase + 3) * 256 + e];
                gv[g] = s + xp[((size_t)b * TT + t) * G4 + g * HH + j0 + jj];
            }
            const float ig = sigmoidf_dev(gv[0]);   // gate order i,f,g,o
            const float fg = sigmoidf_dev(gv[1]);
            const float gg = tanhf(gv[2]);
            const float og = sigmoidf_dev(gv[3]);
            const int cj = b * HH + j0 + jj;
            const float cn = fg * c[cj] + ig * gg;
            const float hn = og * tanhf(cn);
            c[cj] = cn;
            // store h directly as split bf16 for next step's A fragments
            unsigned short hbits = bf16_rne_bits(hn);
            float hf = __builtin_bit_cast(float, (unsigned)hbits << 16);
            h_hi[cj] = __builtin_bit_cast(__bf16, hbits);
            h_lo[cj] = __builtin_bit_cast(__bf16, bf16_rne_bits(hn - hf));
            out[((size_t)b * TT + t) * HH + j0 + jj] = hn;
        }
        grid_barrier(counter, ++epoch, REC_BLOCKS);
    }
}

// ---------------------------------------------------------------------------
extern "C" void kernel_launch(void* const* d_in, const int* in_sizes, int n_in,
                              void* d_out, int out_size, void* d_ws, size_t ws_size,
                              hipStream_t stream) {
    (void)in_sizes; (void)n_in; (void)out_size; (void)ws_size;
    const float* x    = (const float*)d_in[0];
    const float* w_ih = (const float*)d_in[1];
    const float* w_hh = (const float*)d_in[2];
    const float* b_ih = (const float*)d_in[3];
    const float* b_hh = (const float*)d_in[4];
    float* out = (float*)d_out;

    // Workspace layout (all section sizes 64B-multiples):
    char* p = (char*)d_ws;
    float* xp       = (float*)p;            p += (size_t)BB * TT * G4 * 4;   // 512 MB
    float* cbuf     = (float*)p;            p += (size_t)BB * HH * 4;        // 128 KB
    __bf16* h_hi    = (__bf16*)p;           p += (size_t)BB * HH * 2;        //  64 KB
    __bf16* h_lo    = (__bf16*)p;           p += (size_t)BB * HH * 2;
    __bf16* wih_hi  = (__bf16*)p;           p += (size_t)G4 * II * 2;        //   8 MB
    __bf16* wih_lo  = (__bf16*)p;           p += (size_t)G4 * II * 2;
    __bf16* whh_hi  = (__bf16*)p;           p += (size_t)G4 * HH * 2;
    __bf16* whh_lo  = (__bf16*)p;           p += (size_t)G4 * HH * 2;
    __bf16* x_hi    = (__bf16*)p;           p += (size_t)BB * TT * II * 2;   //  64 MB
    __bf16* x_lo    = (__bf16*)p;           p += (size_t)BB * TT * II * 2;
    unsigned* ctr   = (unsigned*)p;

    // One-time splits (memory-bound, ~300 MB total traffic ~ 13 us @ 23.3 TB/s)
    {
        int n8 = (G4 * II) / 8;                          // 524288
        split_kernel<<<(n8 + THREADS - 1) / THREADS, THREADS, 0, stream>>>(w_ih, wih_hi, wih_lo, n8);
        split_kernel<<<(n8 + THREADS - 1) / THREADS, THREADS, 0, stream>>>(w_hh, whh_hi, whh_lo, n8);
        int nx8 = (int)(((size_t)BB * TT * II) / 8);     // 4194304
        split_kernel<<<(nx8 + THREADS - 1) / THREADS, THREADS, 0, stream>>>(x, x_hi, x_lo, nx8);
    }

    lstm_init<<<(BB * HH + THREADS - 1) / THREADS, THREADS, 0, stream>>>(cbuf, h_hi, h_lo, ctr);

    dim3 g1(G4 / 64, (BB * TT) / 128);
    xproj_kernel<<<g1, THREADS, 0, stream>>>(x_hi, x_lo, wih_hi, wih_lo, b_ih, b_hh, xp);

    lstm_recurrent<<<REC_BLOCKS, REC_THREADS, 0, stream>>>(xp, whh_hi, whh_lo, out,
                                                           h_hi, h_lo, cbuf, ctr);
}
